// SingleKeyAttention_56487409877244
// MI455X (gfx1250) — matrix operand order repaired and checked by reading, not compiled, not run on hardware
//
#include <hip/hip_runtime.h>

typedef __attribute__((ext_vector_type(16))) _Float16 v16h;
typedef __attribute__((ext_vector_type(8)))  float    v8f;

#define B_   16
#define D_   256
#define K_   18
#define SZ   128
#define NPOS (SZ * SZ)   // 16384 spatial positions
#define NS   8192        // sampled positions (SIZE*SIZE*0.5)

// ---------------------------------------------------------------------------
// ws layout: [0 .. B_*NPOS) int   : idx_full[b][pos]  (-1 = not sampled)
//            then B_*K_ float     : p2[b][k] = ||prototype||^2
// ---------------------------------------------------------------------------

__global__ void vq_init_idx_kernel(int* __restrict__ idx_full) {
  int i = blockIdx.x * blockDim.x + threadIdx.x;
  if (i < B_ * NPOS) idx_full[i] = -1;
}

__global__ void vq_p2_kernel(const float* __restrict__ protos, float* __restrict__ p2) {
  int i = blockIdx.x * blockDim.x + threadIdx.x;
  if (i < B_ * K_) {
    const float* p = protos + (size_t)i * D_;
    float s = 0.f;
    for (int d = 0; d < D_; ++d) s = fmaf(p[d], p[d], s);
    p2[i] = s;
  }
}

// One wave32 per 16-sample tile. cross = F(16x256) * P^T(256x18->32) via
// 8 steps of v_wmma_f32_16x16x32_f16 over two k-tiles, then argmin of
// p2[k] - 2*cross across lanes.
__global__ void __launch_bounds__(256)
vq_argmin_wmma_kernel(const float* __restrict__ assp,
                      const float* __restrict__ protos,
                      const int*   __restrict__ rnd,
                      const float* __restrict__ p2,
                      int*         __restrict__ idx_full) {
  const int lane = threadIdx.x & 31;
  const int wid  = blockIdx.x * (blockDim.x >> 5) + (threadIdx.x >> 5);
  const int b    = wid >> 9;            // NS/16 = 512 tiles per batch
  const int tile = wid & 511;
  const int base = tile * 16;
  const int r    = lane & 15;
  const int h    = lane >> 4;

  // A operand gather base: sample row M = base + r (both lane halves)
  const int pos = rnd[base + r];
  const float* fptr = assp + (size_t)b * D_ * NPOS + pos;

  // B/C column within tile: N = lane & 15
  const int k0 = r;           // k-tile 0 column (always < 18)
  const int k1 = 16 + r;      // k-tile 1 column (valid only for r < 2)

  v8f acc0 = {};              // cross for k = 0..15
  v8f acc1 = {};              // cross for k = 16..31 (18..31 padded zero)

  for (int d0 = 0; d0 < D_; d0 += 32) {
    // --- A fragment: contiguous k order used here (same as B): K = 16*h + e
    v16h af;
#pragma unroll
    for (int e = 0; e < 16; ++e) {
      const int kk = e + (h << 4);
      af[e] = (_Float16)fptr[(size_t)(d0 + kk) * NPOS];
    }
    // --- B fragments: contiguous k order used here (same as A): row K = 16*h + e, col N = lane&15
    v16h bf0, bf1;
    {
      const float* pp = protos + ((size_t)(b * K_ + k0)) * D_ + d0 + (h << 4);
#pragma unroll
      for (int e = 0; e < 16; ++e) bf0[e] = (_Float16)pp[e];
    }
    if (k1 < K_) {
      const float* pp = protos + ((size_t)(b * K_ + k1)) * D_ + d0 + (h << 4);
#pragma unroll
      for (int e = 0; e < 16; ++e) bf1[e] = (_Float16)pp[e];
    } else {
#pragma unroll
      for (int e = 0; e < 16; ++e) bf1[e] = (_Float16)0.f;
    }

    acc0 = __builtin_amdgcn_wmma_f32_16x16x32_f16(false, af, false, bf0,
                                                  (short)0, acc0, false, false);
    acc1 = __builtin_amdgcn_wmma_f32_16x16x32_f16(false, af, false, bf1,
                                                  (short)0, acc1, false, false);
  }

  // C/D layout: lane holds column k = lane&15; VGPR j holds row M = 8*h + j.
  const float p2a = p2[b * K_ + k0];
  const float p2b = (k1 < K_) ? p2[b * K_ + k1] : 0.f;
  const float INF = __builtin_inff();

#pragma unroll
  for (int j = 0; j < 8; ++j) {
    float s0 = fmaf(-2.f, acc0[j], p2a);
    float s1 = (k1 < K_) ? fmaf(-2.f, acc1[j], p2b) : INF;
    float best = s0; int bk = k0;
    if (s1 < best) { best = s1; bk = k1; }     // k1 > k0, strict < keeps low-k tie-break
#pragma unroll
    for (int m = 1; m < 16; m <<= 1) {         // butterfly within 16-lane half
      float ob = __shfl_xor(best, m, 32);
      int   ok = __shfl_xor(bk,   m, 32);
      if (ob < best || (ob == best && ok < bk)) { best = ob; bk = ok; }
    }
    if (r == 0) {
      const int samp = base + (h << 3) + j;    // sample row M = 8*h + j
      const int sp   = rnd[samp];
      idx_full[b * NPOS + sp] = bk;
    }
  }
}

// Single memory-optimal pass: out = in, substituting prototype vectors at
// sampled positions. float4 / int4 vectorized (pos stays aligned mod 4).
__global__ void __launch_bounds__(256)
vq_copy_subst_kernel(const float* __restrict__ in,
                     const float* __restrict__ protos,
                     const int*   __restrict__ idx_full,
                     float*       __restrict__ out) {
  const size_t t = (size_t)blockIdx.x * blockDim.x + threadIdx.x;
  const size_t e = t * 4;                      // element index, exact cover
  const int pos = (int)(e & (NPOS - 1));
  const int bd  = (int)(e >> 14);
  const int d   = bd & (D_ - 1);
  const int b   = bd >> 8;

  float4 v = *(const float4*)(in + e);
  const int4 ki = *(const int4*)(idx_full + (size_t)b * NPOS + pos);
  const float* pb = protos + (size_t)b * K_ * D_;
  if (ki.x >= 0) v.x = pb[(size_t)ki.x * D_ + d];
  if (ki.y >= 0) v.y = pb[(size_t)ki.y * D_ + d];
  if (ki.z >= 0) v.z = pb[(size_t)ki.z * D_ + d];
  if (ki.w >= 0) v.w = pb[(size_t)ki.w * D_ + d];
  *(float4*)(out + e) = v;
}

extern "C" void kernel_launch(void* const* d_in, const int* in_sizes, int n_in,
                              void* d_out, int out_size, void* d_ws, size_t ws_size,
                              hipStream_t stream) {
  const float* assp   = (const float*)d_in[0];   // (16,256,128,128) f32
  const float* protos = (const float*)d_in[1];   // (16,18,256) f32
  const int*   rnd    = (const int*)d_in[2];     // (8192,) int
  float*       out    = (float*)d_out;

  int*   idx_full = (int*)d_ws;
  float* p2       = (float*)((char*)d_ws + (size_t)B_ * NPOS * sizeof(int));

  vq_init_idx_kernel<<<(B_ * NPOS + 255) / 256, 256, 0, stream>>>(idx_full);
  vq_p2_kernel<<<(B_ * K_ + 63) / 64, 64, 0, stream>>>(protos, p2);

  // B_ * NS/16 = 8192 waves, 8 waves (256 threads) per block -> 1024 blocks
  vq_argmin_wmma_kernel<<<(B_ * (NS / 16)) / 8, 256, 0, stream>>>(
      assp, protos, rnd, p2, idx_full);

  // 67,108,864 elements / 4 per thread / 256 per block = 65536 blocks
  vq_copy_subst_kernel<<<(B_ * D_ * NPOS / 4) / 256, 256, 0, stream>>>(
      assp, protos, idx_full, out);
}
